// Attention_28905129902634
// MI455X (gfx1250) — matrix operand order repaired and checked by reading, not compiled, not run on hardware
//
#include <hip/hip_runtime.h>
#include <hip/hip_bf16.h>

typedef __attribute__((ext_vector_type(16))) __bf16 v16bf;
typedef __attribute__((ext_vector_type(8)))  __bf16 v8bf;
typedef __attribute__((ext_vector_type(4)))  __bf16 v4bf;
typedef __attribute__((ext_vector_type(8)))  float  v8f;
typedef __attribute__((ext_vector_type(4)))  unsigned int u32x4;
typedef __attribute__((ext_vector_type(8)))  unsigned int u32x8;

#define C_DIM 1024
#define NH    16
#define HD    64
#define B_SZ  2
#define T_SZ  2048
#define BT    (B_SZ * T_SZ)   // 4096 rows total

// ---------------------------------------------------------------------------
// Tensor Data Mover: 2D tile load Global -> LDS (CDNA5 TDM, VIMAGE encoding).
// Descriptor packing per ISA cdna5 §8.3/8.4. 2-group form (<=2D tensor).
// LDS receives the tile contiguously row-major: lds[row*tile0 + x].
// Wave-level DMA: issue from ONE wave only; EXEC ignored. Tracked: TENSORcnt.
// ---------------------------------------------------------------------------
__device__ __forceinline__ void tdm_load_2d(unsigned lds_off,
                                            const void* gaddr,
                                            unsigned tensor_dim0,   // elems
                                            unsigned tile0,         // elems
                                            unsigned tile1,         // rows
                                            unsigned stride0)       // elems
{
  unsigned long long ga = (unsigned long long)gaddr;
  u32x4 g0;
  g0[0] = 1u;                                   // count=1, user descriptor
  g0[1] = lds_off;                              // lds_addr (bytes)
  g0[2] = (unsigned)ga;                         // global_addr[31:0]
  g0[3] = (unsigned)(ga >> 32) | (2u << 30);    // global_addr[56:32], type=2
  const unsigned dim1 = 1u << 20;               // generous (no OOB in our use)
  u32x8 g1;
  g1[0] = (1u << 16);                           // data_size=1 (2 bytes)
  g1[1] = (tensor_dim0 & 0xFFFFu) << 16;        // tensor_dim0[15:0]
  g1[2] = (tensor_dim0 >> 16) | ((dim1 & 0xFFFFu) << 16);
  g1[3] = (dim1 >> 16) | (tile0 << 16);         // tensor_dim1[31:16], tile_dim0
  g1[4] = tile1;                                // tile_dim1 (tile_dim2 = 0)
  g1[5] = stride0;                              // tensor_dim0_stride[31:0]
  g1[6] = 0u;
  g1[7] = 0u;
  asm volatile("tensor_load_to_lds %0, %1" :: "s"(g0), "s"(g1) : "memory");
}

__device__ __forceinline__ unsigned lds_offset_of(const void* p) {
  // Generic LDS pointer = {shared aperture, offset[31:0]} -> truncate.
  return (unsigned)(unsigned long long)p;
}

// ---------------------------------------------------------------------------
// fp32 -> bf16 conversion (vectorized float4 -> 4x bf16)
// ---------------------------------------------------------------------------
__global__ void f32_to_bf16_kernel(const float* __restrict__ in,
                                   __bf16* __restrict__ out, int n4) {
  int i = blockIdx.x * blockDim.x + threadIdx.x;
  if (i < n4) {
    float4 v = ((const float4*)in)[i];
    v4bf o;
    o.x = (__bf16)v.x; o.y = (__bf16)v.y; o.z = (__bf16)v.z; o.w = (__bf16)v.w;
    ((v4bf*)out)[i] = o;
  }
}

// ---------------------------------------------------------------------------
// GEMM: out[m,n] = sum_k A[m,k] * W[n,k]   (y = A @ W^T)
// A: [M x K] bf16 row-major.  W: [N x K] bf16 row-major (nn.Linear layout).
// Block 256 threads = 8 waves; block tile 128(M) x 128(N); wave tile 32x64.
// K-step 64. Tiles staged by TDM (tensor_load_to_lds), double-buffered so
// DMA of tile i+1 overlaps WMMA on tile i. 16 WMMAs / wave / K-step.
// ---------------------------------------------------------------------------
template<bool STORE_BF16>
__global__ __launch_bounds__(256) void gemm_xwT_kernel(
    const __bf16* __restrict__ A, const __bf16* __restrict__ W,
    __bf16* __restrict__ outB, float* __restrict__ outF,
    int M, int N, int K)
{
  __shared__ __bf16 As[2][128 * 64];
  __shared__ __bf16 Bs[2][128 * 64];

  const int tid  = threadIdx.x;
  const int lane = tid & 31;
  const int wave = tid >> 5;
  const int wm   = wave & 3;   // M sub-tile (32 rows each)
  const int wn   = wave >> 2;  // N sub-tile (64 cols each)
  const int m0   = blockIdx.y * 128;
  const int n0   = blockIdx.x * 128;

  v8f acc[2][4] = {};

  const int arow = lane & 15;          // A-frag row within 16
  const int ak   = (lane >> 4) * 16;   // A-frag K-half select
  const int bk   = (lane >> 4) * 16;   // B-frag K-half select

  const int n_iter = K / 64;

  // prologue: DMA tile 0 into buffer 0
  if (wave == 0) {
    tdm_load_2d(lds_offset_of(&As[0][0]), &A[(size_t)m0 * K], K, 64, 128, K);
    tdm_load_2d(lds_offset_of(&Bs[0][0]), &W[(size_t)n0 * K], K, 64, 128, K);
    __builtin_amdgcn_s_wait_tensorcnt(0);
  }
  __syncthreads();

  for (int it = 0; it < n_iter; ++it) {
    const int buf = it & 1;
    // kick DMA for next K-tile into the other buffer (overlaps compute)
    if (it + 1 < n_iter && wave == 0) {
      const int k1 = (it + 1) * 64;
      tdm_load_2d(lds_offset_of(&As[buf ^ 1][0]), &A[(size_t)m0 * K + k1],
                  K, 64, 128, K);
      tdm_load_2d(lds_offset_of(&Bs[buf ^ 1][0]), &W[(size_t)n0 * K + k1],
                  K, 64, 128, K);
    }

    // ---- A fragments: 2 M-subtiles x 2 K-chunks (16x32 each)
    v16bf af[2][2];
    #pragma unroll
    for (int ms = 0; ms < 2; ++ms) {
      #pragma unroll
      for (int c = 0; c < 2; ++c) {
        const __bf16* p = &As[buf][(wm * 32 + ms * 16 + arow) * 64 + c * 32 + ak];
        v8bf lo = *(const v8bf*)p;
        v8bf hi = *(const v8bf*)(p + 8);
        #pragma unroll
        for (int i = 0; i < 8; ++i) { af[ms][c][i] = lo[i]; af[ms][c][8 + i] = hi[i]; }
      }
    }

    // ---- B fragments on the fly: 4 N-subtiles x 2 K-chunks, 16 WMMAs
    #pragma unroll
    for (int s = 0; s < 4; ++s) {
      int bn = (wn * 64 + s * 16 + (lane & 15)) * 64;
      #pragma unroll
      for (int c = 0; c < 2; ++c) {
        v16bf bfrag = *(const v16bf*)&Bs[buf][bn + c * 32 + bk];
        #pragma unroll
        for (int ms = 0; ms < 2; ++ms) {
          acc[ms][s] = __builtin_amdgcn_wmma_f32_16x16x32_bf16(
              false, af[ms][c], false, bfrag, (short)0, acc[ms][s], false, false);
        }
      }
    }

    if (wave == 0) __builtin_amdgcn_s_wait_tensorcnt(0);
    __syncthreads();
  }

  // ---- epilogue (C layout: M = r + 8*(lane>=16), N = lane&15)
  #pragma unroll
  for (int ms = 0; ms < 2; ++ms) {
    #pragma unroll
    for (int s = 0; s < 4; ++s) {
      #pragma unroll
      for (int r = 0; r < 8; ++r) {
        int m = m0 + wm * 32 + ms * 16 + r + 8 * (lane >> 4);
        int n = n0 + wn * 64 + s * 16 + (lane & 15);
        if (STORE_BF16) outB[(size_t)m * N + n] = (__bf16)acc[ms][s][r];
        else            outF[(size_t)m * N + n] = acc[ms][s][r];
      }
    }
  }
}

// ---------------------------------------------------------------------------
// Fused causal flash-attention. Q/K/V/O are bf16 [B*T, C]; head h lives in
// columns h*64..h*64+63. Block = 128 threads (4 waves) = 64 queries;
// each wave owns 16 queries. Loop over 64-key tiles up to causal diagonal.
// K tile staged by TDM; V tile transposed manually into LDS.
// ---------------------------------------------------------------------------
__global__ __launch_bounds__(128) void attn_fused_kernel(
    const __bf16* __restrict__ Q, const __bf16* __restrict__ K,
    const __bf16* __restrict__ V, __bf16* __restrict__ O)
{
  __shared__ __bf16 Ks[64 * 64];       // [key][d]  (TDM destination)
  __shared__ __bf16 Vs[64 * 64];       // transposed: [d][key]
  __shared__ __bf16 Ps[4 * 16 * 64];   // per-wave P tile [q][key]

  const int tid  = threadIdx.x;
  const int lane = tid & 31;
  const int wave = tid >> 5;
  const int qt   = blockIdx.x;   // query tile (64 queries)
  const int h    = blockIdx.y;
  const int b    = blockIdx.z;

  const size_t base  = (size_t)b * T_SZ * C_DIM + (size_t)h * HD;
  const int    qbase = qt * 64 + wave * 16;

  const int arow = lane & 15;
  const int ak   = (lane >> 4) * 16;
  const int bk   = (lane >> 4) * 16;

  // ---- load Q fragments (two 16x32 A-frags covering d=0..63)
  v16bf qfrag[2];
  #pragma unroll
  for (int c = 0; c < 2; ++c) {
    const __bf16* qp = &Q[base + (size_t)(qbase + arow) * C_DIM + c * 32 + ak];
    v8bf lo = *(const v8bf*)qp;
    v8bf hi = *(const v8bf*)(qp + 8);
    #pragma unroll
    for (int i = 0; i < 8; ++i) { qfrag[c][i] = lo[i]; qfrag[c][8 + i] = hi[i]; }
  }

  v8f oacc[4] = {};
  float mrun[8], lrun[8];
  #pragma unroll
  for (int r = 0; r < 8; ++r) { mrun[r] = -1e30f; lrun[r] = 0.f; }

  for (int kt = 0; kt <= qt; ++kt) {
    // ---- TDM: K tile 64x64 (row pitch C_DIM) -> Ks, issued by wave 0
    if (wave == 0) {
      tdm_load_2d(lds_offset_of(&Ks[0]),
                  &K[base + (size_t)(kt * 64) * C_DIM],
                  C_DIM, 64, 64, C_DIM);
    }
    // ---- V tile: manual transposed stage (TDM cannot transpose)
    #pragma unroll
    for (int i = 0; i < 4; ++i) {
      int c   = tid + i * 128;
      int row = c >> 3;          // key within tile
      int cc  = (c & 7) * 8;     // d chunk
      uint4 vv = *(const uint4*)&V[base + (size_t)(kt * 64 + row) * C_DIM + cc];
      const __bf16* ve = (const __bf16*)&vv;
      #pragma unroll
      for (int e = 0; e < 8; ++e) Vs[(cc + e) * 64 + row] = ve[e];
    }
    if (wave == 0) __builtin_amdgcn_s_wait_tensorcnt(0);
    __syncthreads();

    // ---- S = Q @ K^T  (4 key sub-tiles of 16, each needs 2 WMMAs over d)
    v8f sacc[4];
    #pragma unroll
    for (int s = 0; s < 4; ++s) {
      int key = s * 16 + (lane & 15);
      v16bf b0 = *(const v16bf*)&Ks[key * 64 + bk];
      v16bf b1 = *(const v16bf*)&Ks[key * 64 + 32 + bk];
      v8f z = {};
      z = __builtin_amdgcn_wmma_f32_16x16x32_bf16(false, qfrag[0], false, b0,
                                                  (short)0, z, false, false);
      z = __builtin_amdgcn_wmma_f32_16x16x32_bf16(false, qfrag[1], false, b1,
                                                  (short)0, z, false, false);
      sacc[s] = z;
    }

    // ---- scale + causal mask
    #pragma unroll
    for (int s = 0; s < 4; ++s) {
      int kg = kt * 64 + s * 16 + (lane & 15);
      #pragma unroll
      for (int r = 0; r < 8; ++r) {
        int qg = qbase + r + 8 * (lane >> 4);
        float sv = sacc[s][r] * 0.125f;   // 1/sqrt(64)
        sacc[s][r] = (kg <= qg) ? sv : -1e30f;
      }
    }

    // ---- online softmax: row max / rescale / exp / row sum
    float mnew[8], alpha[8];
    #pragma unroll
    for (int r = 0; r < 8; ++r) {
      float mx = fmaxf(fmaxf(sacc[0][r], sacc[1][r]),
                       fmaxf(sacc[2][r], sacc[3][r]));
      #pragma unroll
      for (int d = 1; d < 16; d <<= 1) mx = fmaxf(mx, __shfl_xor(mx, d, 32));
      mnew[r]  = fmaxf(mrun[r], mx);
      alpha[r] = __expf(mrun[r] - mnew[r]);
      mrun[r]  = mnew[r];
    }

    #pragma unroll
    for (int r = 0; r < 8; ++r) {
      float rs = 0.f;
      int prow = r + 8 * (lane >> 4);
      #pragma unroll
      for (int s = 0; s < 4; ++s) {
        float p = __expf(sacc[s][r] - mnew[r]);
        rs += p;
        Ps[wave * (16 * 64) + prow * 64 + s * 16 + (lane & 15)] = (__bf16)p;
      }
      #pragma unroll
      for (int d = 1; d < 16; d <<= 1) rs += __shfl_xor(rs, d, 32);
      lrun[r] = lrun[r] * alpha[r] + rs;
      #pragma unroll
      for (int ds = 0; ds < 4; ++ds) oacc[ds][r] *= alpha[r];
    }

    // ---- reload P as A-fragments (C-layout -> A-layout via LDS)
    v16bf pfrag[2];
    #pragma unroll
    for (int c = 0; c < 2; ++c) {
      const __bf16* pp = &Ps[wave * (16 * 64) + arow * 64 + c * 32 + ak];
      v8bf lo = *(const v8bf*)pp;
      v8bf hi = *(const v8bf*)(pp + 8);
      #pragma unroll
      for (int i = 0; i < 8; ++i) { pfrag[c][i] = lo[i]; pfrag[c][8 + i] = hi[i]; }
    }

    // ---- O += P @ V  (4 d sub-tiles x 2 key chunks)
    #pragma unroll
    for (int ds = 0; ds < 4; ++ds) {
      int dn = ds * 16 + (lane & 15);
      v16bf bv0 = *(const v16bf*)&Vs[dn * 64 + bk];
      v16bf bv1 = *(const v16bf*)&Vs[dn * 64 + 32 + bk];
      oacc[ds] = __builtin_amdgcn_wmma_f32_16x16x32_bf16(
          false, pfrag[0], false, bv0, (short)0, oacc[ds], false, false);
      oacc[ds] = __builtin_amdgcn_wmma_f32_16x16x32_bf16(
          false, pfrag[1], false, bv1, (short)0, oacc[ds], false, false);
    }
    __syncthreads();
  }

  // ---- normalize and store O (bf16)
  #pragma unroll
  for (int r = 0; r < 8; ++r) {
    float inv = 1.0f / lrun[r];
    int qg = qbase + r + 8 * (lane >> 4);
    #pragma unroll
    for (int ds = 0; ds < 4; ++ds) {
      int dn = ds * 16 + (lane & 15);
      O[base + (size_t)qg * C_DIM + dn] = (__bf16)(oacc[ds][r] * inv);
    }
  }
}

// ---------------------------------------------------------------------------
// Host-side launcher
// ---------------------------------------------------------------------------
extern "C" void kernel_launch(void* const* d_in, const int* in_sizes, int n_in,
                              void* d_out, int out_size, void* d_ws, size_t ws_size,
                              hipStream_t stream) {
  const float* x  = (const float*)d_in[0];
  const float* Wq = (const float*)d_in[1];
  const float* Wk = (const float*)d_in[2];
  const float* Wv = (const float*)d_in[3];
  const float* Wo = (const float*)d_in[4];
  float* out = (float*)d_out;

  char* ws = (char*)d_ws;
  const size_t MiB = 1u << 20;
  __bf16* xb  = (__bf16*)(ws);               // 8 MiB, reused for O after QKV
  __bf16* ob  = xb;
  __bf16* wqb = (__bf16*)(ws +  8 * MiB);
  __bf16* wkb = (__bf16*)(ws + 10 * MiB);
  __bf16* wvb = (__bf16*)(ws + 12 * MiB);
  __bf16* wob = (__bf16*)(ws + 14 * MiB);
  __bf16* qb  = (__bf16*)(ws + 16 * MiB);
  __bf16* kb  = (__bf16*)(ws + 24 * MiB);
  __bf16* vb  = (__bf16*)(ws + 32 * MiB);

  // fp32 -> bf16 conversions
  {
    int nx4 = (BT * C_DIM) / 4;
    int nw4 = (C_DIM * C_DIM) / 4;
    f32_to_bf16_kernel<<<(nx4 + 255) / 256, 256, 0, stream>>>(x,  xb,  nx4);
    f32_to_bf16_kernel<<<(nw4 + 255) / 256, 256, 0, stream>>>(Wq, wqb, nw4);
    f32_to_bf16_kernel<<<(nw4 + 255) / 256, 256, 0, stream>>>(Wk, wkb, nw4);
    f32_to_bf16_kernel<<<(nw4 + 255) / 256, 256, 0, stream>>>(Wv, wvb, nw4);
    f32_to_bf16_kernel<<<(nw4 + 255) / 256, 256, 0, stream>>>(Wo, wob, nw4);
  }

  // QKV projections: [4096 x 1024] @ [1024 x 1024]^T, 128x128 block tiles
  dim3 ggrid(C_DIM / 128, BT / 128);
  gemm_xwT_kernel<true><<<ggrid, 256, 0, stream>>>(xb, wqb, qb, nullptr,
                                                   BT, C_DIM, C_DIM);
  gemm_xwT_kernel<true><<<ggrid, 256, 0, stream>>>(xb, wkb, kb, nullptr,
                                                   BT, C_DIM, C_DIM);
  gemm_xwT_kernel<true><<<ggrid, 256, 0, stream>>>(xb, wvb, vb, nullptr,
                                                   BT, C_DIM, C_DIM);

  // fused causal attention (writes ob, which aliases xb -- x no longer needed)
  attn_fused_kernel<<<dim3(T_SZ / 64, NH, B_SZ), 128, 0, stream>>>(qb, kb, vb, ob);

  // output projection -> fp32 d_out
  gemm_xwT_kernel<false><<<ggrid, 256, 0, stream>>>(ob, wob, nullptr, out,
                                                    BT, C_DIM, C_DIM);
}
